// LSTMAttentionDecoder_65326452572824
// MI455X (gfx1250) — matrix operand order repaired and checked by reading, not compiled, not run on hardware
//
#include <hip/hip_runtime.h>

// ---------------------------------------------------------------------------
// LSTM + global attention decoder for MI455X (gfx1250), bf16 WMMA everywhere.
// B=64, T=128, S=256, H=I=1024.
// ---------------------------------------------------------------------------

#define Bsz 64
#define Tsz 128
#define Ssz 256
#define Hsz 1024
#define G4H 4096

typedef unsigned short ushort_t;
typedef __attribute__((ext_vector_type(16))) __bf16 v16bf;
typedef __attribute__((ext_vector_type(8)))  float  v8f;

union FragU {
  v16bf v;
  uint4 q[2];
};

__device__ __forceinline__ ushort_t f2bf(float x) {
  unsigned u = __builtin_bit_cast(unsigned, x);
  unsigned r = u + 0x7FFFu + ((u >> 16) & 1u);
  return (ushort_t)(r >> 16);
}

__device__ __forceinline__ v8f wmma_bf16(v16bf a, v16bf b, v8f c) {
  return __builtin_amdgcn_wmma_f32_16x16x32_bf16(
      false, a, false, b, (short)0, c, false, false);
}

// A-operand fragment: 16x32 (MxK), rows row0..row0+15, K k0..k0+31.
// Lanes 0-15 hold K {k0..k0+15}; lanes 16-31 hold K {k0+16..k0+31}.
__device__ __forceinline__ v16bf load_a(const ushort_t* m, int ld, int row0, int k0) {
  int lane = threadIdx.x & 31;
  const ushort_t* p = m + (long)(row0 + (lane & 15)) * ld + k0 + ((lane >> 4) << 4);
  FragU f;
  f.q[0] = *(const uint4*)(p);
  f.q[1] = *(const uint4*)(p + 8);
  return f.v;
}

// B-operand fragment: 32x16 (KxN) sourced from a row-major [N,K] matrix.
// Lane n (0-15) holds column n with K {k0..k0+15}; lanes 16-31 hold K {k0+16..k0+31}.
__device__ __forceinline__ v16bf load_b(const ushort_t* m, int ld, int col0, int k0) {
  int lane = threadIdx.x & 31;
  const ushort_t* p = m + (long)(col0 + (lane & 15)) * ld + k0 + ((lane >> 4) << 4);
  FragU f;
  f.q[0] = *(const uint4*)(p);
  f.q[1] = *(const uint4*)(p + 8);
  return f.v;
}

// ---------------------------------------------------------------------------
// elementwise helpers
// ---------------------------------------------------------------------------
__global__ void k_cvt_bf16(const float* __restrict__ in, ushort_t* __restrict__ out, long n) {
  long i = (long)blockIdx.x * blockDim.x + threadIdx.x;
  if (i < n) out[i] = f2bf(in[i]);
}

__global__ void k_bias_sum(const float* __restrict__ a, const float* __restrict__ b,
                           float* __restrict__ out, int n) {
  int i = blockIdx.x * blockDim.x + threadIdx.x;
  if (i < n) out[i] = a[i] + b[i];
}

__global__ void k_init_state(const float* __restrict__ h0, const float* __restrict__ c0,
                             float* __restrict__ h_st, float* __restrict__ c_st,
                             ushort_t* __restrict__ h_bf0, int n) {
  int i = blockIdx.x * blockDim.x + threadIdx.x;
  if (i < n) {
    float h = h0[i];
    h_st[i] = h;
    c_st[i] = c0[i];
    h_bf0[i] = f2bf(h);
  }
}

// ctx [B,S,H] f32 -> ctxT [B,H,S] bf16, LDS-tiled transpose.
__global__ void k_ctx_transpose(const float* __restrict__ ctx, ushort_t* __restrict__ ctxT) {
  __shared__ ushort_t tile[32][33];
  int b = blockIdx.z;
  int s0 = blockIdx.x * 32;
  int h0 = blockIdx.y * 32;
  int tx = threadIdx.x, ty = threadIdx.y;  // 32 x 8
  const float* src = ctx + ((long)b * Ssz + s0) * Hsz + h0;
  for (int i = 0; i < 32; i += 8)
    tile[ty + i][tx] = f2bf(src[(long)(ty + i) * Hsz + tx]);
  __syncthreads();
  ushort_t* dst = ctxT + ((long)b * Hsz + h0) * Ssz + s0;
  for (int i = 0; i < 32; i += 8)
    dst[(long)(ty + i) * Ssz + tx] = tile[tx][ty + i];
}

// ---------------------------------------------------------------------------
// Generic bf16 GEMM: Y[M,N] = A[M,K] @ B[N,K]^T  (+bias), fp32 or bf16 out.
// Block = 256 threads = 8 waves; block tile 128(M) x 128(N);
// wave tile 32(M) x 64(N) = 2 A-frags x 4 B-frags -> 8 WMMA per K-step.
// grid = (M/128, N/128, batch)
// ---------------------------------------------------------------------------
__global__ void k_gemm_bf16(const ushort_t* __restrict__ A, const ushort_t* __restrict__ B,
                            const float* __restrict__ bias,
                            float* __restrict__ Cf, ushort_t* __restrict__ Cb,
                            int K, int lda, int ldb, int ldc,
                            long sA, long sB, long sC) {
  int z = blockIdx.z;
  A += (long)z * sA;
  B += (long)z * sB;

  int wid = threadIdx.x >> 5;
  int waveM = wid & 3;   // 4 waves along M
  int waveN = wid >> 2;  // 2 waves along N
  int rowBase = blockIdx.x * 128 + waveM * 32;
  int colBase = blockIdx.y * 128 + waveN * 64;
  int lane = threadIdx.x & 31;
  int half = lane >> 4;
  int nl = lane & 15;

  v8f acc[2][4] = {};
  for (int k = 0; k < K; k += 32) {
    v16bf a0 = load_a(A, lda, rowBase, k);
    v16bf a1 = load_a(A, lda, rowBase + 16, k);
    v16bf b0 = load_b(B, ldb, colBase, k);
    v16bf b1 = load_b(B, ldb, colBase + 16, k);
    v16bf b2 = load_b(B, ldb, colBase + 32, k);
    v16bf b3 = load_b(B, ldb, colBase + 48, k);
    acc[0][0] = wmma_bf16(a0, b0, acc[0][0]);
    acc[0][1] = wmma_bf16(a0, b1, acc[0][1]);
    acc[0][2] = wmma_bf16(a0, b2, acc[0][2]);
    acc[0][3] = wmma_bf16(a0, b3, acc[0][3]);
    acc[1][0] = wmma_bf16(a1, b0, acc[1][0]);
    acc[1][1] = wmma_bf16(a1, b1, acc[1][1]);
    acc[1][2] = wmma_bf16(a1, b2, acc[1][2]);
    acc[1][3] = wmma_bf16(a1, b3, acc[1][3]);
  }

  for (int i = 0; i < 2; ++i)
    for (int j = 0; j < 4; ++j) {
      int c0 = colBase + j * 16 + nl;
      float bv = bias ? bias[c0] : 0.0f;
      for (int r = 0; r < 8; ++r) {
        int row = rowBase + i * 16 + r + 8 * half;
        float v = acc[i][j][r] + bv;
        long idx = (long)z * sC + (long)row * ldc + c0;
        if (Cf) Cf[idx] = v;
        else    Cb[idx] = f2bf(v);
      }
    }
}

// ---------------------------------------------------------------------------
// One LSTM timestep: gates = x_gates[:,t,:] + h @ W_hh^T ; cell update.
// Each wave: one 16-row M-tile (of 64 batch rows), one 16-col hidden tile,
// computing all four gate tiles (i,f,g,o reuse the A fragment).
// grid = 32 blocks x 256 threads = 256 waves = 4 Mtiles * 64 Ntiles.
// ---------------------------------------------------------------------------
__global__ void k_lstm_step(const ushort_t* __restrict__ h_in,   // [64,1024] bf16
                            ushort_t* __restrict__ h_out,        // [64,1024] bf16
                            const ushort_t* __restrict__ Whh,    // [4096,1024] bf16
                            const float* __restrict__ xg,        // [B,T,4096] f32 (incl. bias)
                            float* __restrict__ c_st,            // [64,1024] f32
                            float* __restrict__ h_st,            // [64,1024] f32
                            ushort_t* __restrict__ lstm_bf,      // [B,T,1024] bf16
                            int t) {
  int w = blockIdx.x * 8 + (threadIdx.x >> 5);
  int mtile = w & 3;
  int ntile = w >> 2;
  int row0 = mtile * 16;
  int n0 = ntile * 16;
  int lane = threadIdx.x & 31;
  int half = lane >> 4;
  int nl = lane & 15;

  v8f ai = {}, af = {}, ag = {}, ao = {};
  for (int k = 0; k < Hsz; k += 32) {
    v16bf a = load_a(h_in, Hsz, row0, k);
    v16bf bi = load_b(Whh, Hsz, 0 * Hsz + n0, k);
    v16bf bf = load_b(Whh, Hsz, 1 * Hsz + n0, k);
    v16bf bg = load_b(Whh, Hsz, 2 * Hsz + n0, k);
    v16bf bo = load_b(Whh, Hsz, 3 * Hsz + n0, k);
    ai = wmma_bf16(a, bi, ai);
    af = wmma_bf16(a, bf, af);
    ag = wmma_bf16(a, bg, ag);
    ao = wmma_bf16(a, bo, ao);
  }

  int j = n0 + nl;
  for (int r = 0; r < 8; ++r) {
    int b = row0 + r + 8 * half;
    long xb = ((long)b * Tsz + t) * G4H + j;
    float gi = ai[r] + xg[xb];
    float gf = af[r] + xg[xb + Hsz];
    float gg = ag[r] + xg[xb + 2 * Hsz];
    float go = ao[r] + xg[xb + 3 * Hsz];
    float si = 1.0f / (1.0f + expf(-gi));
    float sf = 1.0f / (1.0f + expf(-gf));
    float tg = tanhf(gg);
    float so = 1.0f / (1.0f + expf(-go));
    long hidx = (long)b * Hsz + j;
    float c = sf * c_st[hidx] + si * tg;
    float h = so * tanhf(c);
    c_st[hidx] = c;
    h_st[hidx] = h;
    ushort_t hb = f2bf(h);
    h_out[hidx] = hb;
    lstm_bf[((long)b * Tsz + t) * Hsz + j] = hb;
  }
}

// ---------------------------------------------------------------------------
// Masked softmax over S=256: scores f32 [B,T,S] -> attn bf16 [B,T,S]
// ---------------------------------------------------------------------------
__global__ void k_softmax(const float* __restrict__ scores, const int* __restrict__ src_len,
                          ushort_t* __restrict__ attn) {
  __shared__ float red[256];
  int row = blockIdx.x;            // b*T + t
  int b = row / Tsz;
  int tid = threadIdx.x;           // s
  bool ok = tid < src_len[b];
  float val = scores[(long)row * Ssz + tid];
  float v = ok ? val : -INFINITY;
  red[tid] = v;
  __syncthreads();
  for (int s = 128; s > 0; s >>= 1) {
    if (tid < s) red[tid] = fmaxf(red[tid], red[tid + s]);
    __syncthreads();
  }
  float mx = red[0];
  __syncthreads();
  float e = ok ? expf(val - mx) : 0.0f;
  red[tid] = e;
  __syncthreads();
  for (int s = 128; s > 0; s >>= 1) {
    if (tid < s) red[tid] += red[tid + s];
    __syncthreads();
  }
  float inv = 1.0f / red[0];
  attn[(long)row * Ssz + tid] = f2bf(e * inv);
}

// ---------------------------------------------------------------------------
// Output projection: h_tilde = tanh([context, lstm_out] @ W_out^T)
// K = 2048 walked in one loop, A switches between context (k<1024) and
// lstm_out (k>=1024); B = W_out row-major [1024, 2048]. fp32 out to d_out.
// grid = (M/128, N/128) = (64, 8); wave tile 32x64.
// ---------------------------------------------------------------------------
__global__ void k_htilde(const ushort_t* __restrict__ ctxv,   // [B*T,1024] bf16
                         const ushort_t* __restrict__ lstm,   // [B*T,1024] bf16
                         const ushort_t* __restrict__ Wout,   // [1024,2048] bf16
                         float* __restrict__ out) {           // [B*T,1024] f32
  int wid = threadIdx.x >> 5;
  int waveM = wid & 3;
  int waveN = wid >> 2;
  int rowBase = blockIdx.x * 128 + waveM * 32;
  int colBase = blockIdx.y * 128 + waveN * 64;
  int lane = threadIdx.x & 31;
  int half = lane >> 4;
  int nl = lane & 15;

  v8f acc[2][4] = {};
  for (int k = 0; k < 2 * Hsz; k += 32) {
    const ushort_t* A = (k < Hsz) ? ctxv : lstm;
    int ka = (k < Hsz) ? k : (k - Hsz);
    v16bf a0 = load_a(A, Hsz, rowBase, ka);
    v16bf a1 = load_a(A, Hsz, rowBase + 16, ka);
    v16bf b0 = load_b(Wout, 2 * Hsz, colBase, k);
    v16bf b1 = load_b(Wout, 2 * Hsz, colBase + 16, k);
    v16bf b2 = load_b(Wout, 2 * Hsz, colBase + 32, k);
    v16bf b3 = load_b(Wout, 2 * Hsz, colBase + 48, k);
    acc[0][0] = wmma_bf16(a0, b0, acc[0][0]);
    acc[0][1] = wmma_bf16(a0, b1, acc[0][1]);
    acc[0][2] = wmma_bf16(a0, b2, acc[0][2]);
    acc[0][3] = wmma_bf16(a0, b3, acc[0][3]);
    acc[1][0] = wmma_bf16(a1, b0, acc[1][0]);
    acc[1][1] = wmma_bf16(a1, b1, acc[1][1]);
    acc[1][2] = wmma_bf16(a1, b2, acc[1][2]);
    acc[1][3] = wmma_bf16(a1, b3, acc[1][3]);
  }

  for (int i = 0; i < 2; ++i)
    for (int j = 0; j < 4; ++j) {
      int c0 = colBase + j * 16 + nl;
      for (int r = 0; r < 8; ++r) {
        int row = rowBase + i * 16 + r + 8 * half;
        out[(long)row * Hsz + c0] = tanhf(acc[i][j][r]);
      }
    }
}

__global__ void k_finalize(const float* __restrict__ h_st, const float* __restrict__ c_st,
                           float* __restrict__ out) {
  int i = blockIdx.x * blockDim.x + threadIdx.x;
  const int n = Bsz * Hsz;
  long base = (long)Bsz * Tsz * Hsz;
  if (i < n) out[base + i] = h_st[i];
  else if (i < 2 * n) out[base + i] = c_st[i - n];
}

// ---------------------------------------------------------------------------
// host side
// ---------------------------------------------------------------------------
extern "C" void kernel_launch(void* const* d_in, const int* in_sizes, int n_in,
                              void* d_out, int out_size, void* d_ws, size_t ws_size,
                              hipStream_t stream) {
  const float* trg   = (const float*)d_in[0];   // [B,T,I]
  const float* h0    = (const float*)d_in[1];   // [B,H]
  const float* c0    = (const float*)d_in[2];   // [B,H]
  const float* ctx   = (const float*)d_in[3];   // [B,S,H]
  const int*   slen  = (const int*)d_in[4];     // [B]
  const float* W_ih  = (const float*)d_in[5];   // [4H,I]
  const float* W_hh  = (const float*)d_in[6];   // [4H,H]
  const float* b_ih  = (const float*)d_in[7];   // [4H]
  const float* b_hh  = (const float*)d_in[8];   // [4H]
  const float* W_in  = (const float*)d_in[9];   // [H,H]
  const float* W_out = (const float*)d_in[10];  // [H,2H]
  float* out = (float*)d_out;

  char* ws = (char*)d_ws;
  size_t off = 0;
  auto alloc = [&](size_t bytes) -> char* {
    char* p = ws + off;
    off += (bytes + 255) & ~(size_t)255;
    return p;
  };

  ushort_t* trg_bf  = (ushort_t*)alloc((size_t)Bsz * Tsz * Hsz * 2);
  ushort_t* Wih_bf  = (ushort_t*)alloc((size_t)G4H * Hsz * 2);
  ushort_t* Whh_bf  = (ushort_t*)alloc((size_t)G4H * Hsz * 2);
  ushort_t* Win_bf  = (ushort_t*)alloc((size_t)Hsz * Hsz * 2);
  ushort_t* Wout_bf = (ushort_t*)alloc((size_t)Hsz * 2 * Hsz * 2);
  ushort_t* ctx_bf  = (ushort_t*)alloc((size_t)Bsz * Ssz * Hsz * 2);
  ushort_t* ctxT_bf = (ushort_t*)alloc((size_t)Bsz * Hsz * Ssz * 2);
  float*    xg      = (float*)alloc((size_t)Bsz * Tsz * G4H * 4);
  float*    bias    = (float*)alloc((size_t)G4H * 4);
  float*    h_st    = (float*)alloc((size_t)Bsz * Hsz * 4);
  float*    c_st    = (float*)alloc((size_t)Bsz * Hsz * 4);
  ushort_t* h_bf    = (ushort_t*)alloc((size_t)2 * Bsz * Hsz * 2);  // double buffer
  ushort_t* lstm_bf = (ushort_t*)alloc((size_t)Bsz * Tsz * Hsz * 2);
  ushort_t* q_bf    = (ushort_t*)alloc((size_t)Bsz * Tsz * Hsz * 2);
  float*    scores  = (float*)alloc((size_t)Bsz * Tsz * Ssz * 4);
  ushort_t* attn_bf = (ushort_t*)alloc((size_t)Bsz * Tsz * Ssz * 2);
  ushort_t* cvec_bf = (ushort_t*)alloc((size_t)Bsz * Tsz * Hsz * 2);
  (void)ws_size; (void)in_sizes; (void)n_in; (void)out_size;

  auto cvt = [&](const float* src, ushort_t* dst, long n) {
    k_cvt_bf16<<<dim3((unsigned)((n + 255) / 256)), 256, 0, stream>>>(src, dst, n);
  };
  cvt(trg,   trg_bf,  (long)Bsz * Tsz * Hsz);
  cvt(W_ih,  Wih_bf,  (long)G4H * Hsz);
  cvt(W_hh,  Whh_bf,  (long)G4H * Hsz);
  cvt(W_in,  Win_bf,  (long)Hsz * Hsz);
  cvt(W_out, Wout_bf, (long)Hsz * 2 * Hsz);
  cvt(ctx,   ctx_bf,  (long)Bsz * Ssz * Hsz);

  k_bias_sum<<<dim3(G4H / 256), 256, 0, stream>>>(b_ih, b_hh, bias, G4H);
  k_init_state<<<dim3(Bsz * Hsz / 256), 256, 0, stream>>>(h0, c0, h_st, c_st, h_bf, Bsz * Hsz);
  k_ctx_transpose<<<dim3(Ssz / 32, Hsz / 32, Bsz), dim3(32, 8), 0, stream>>>(ctx, ctxT_bf);

  // x_gates = trg @ W_ih^T + (b_ih + b_hh)   : M=8192, N=4096, K=1024, f32 out
  k_gemm_bf16<<<dim3(Bsz * Tsz / 128, G4H / 128, 1), 256, 0, stream>>>(
      trg_bf, Wih_bf, bias, xg, nullptr, Hsz, Hsz, Hsz, G4H, 0, 0, 0);

  // recurrence
  for (int t = 0; t < Tsz; ++t) {
    const ushort_t* hin = h_bf + (size_t)(t & 1) * Bsz * Hsz;
    ushort_t* hout = h_bf + (size_t)((t + 1) & 1) * Bsz * Hsz;
    k_lstm_step<<<dim3(32), 256, 0, stream>>>(hin, hout, Whh_bf, xg, c_st, h_st, lstm_bf, t);
  }

  // q = lstm_out @ W_in^T : M=8192, N=1024, K=1024, bf16 out
  k_gemm_bf16<<<dim3(Bsz * Tsz / 128, Hsz / 128, 1), 256, 0, stream>>>(
      lstm_bf, Win_bf, nullptr, nullptr, q_bf, Hsz, Hsz, Hsz, Hsz, 0, 0, 0);

  // scores = q @ ctx^T per batch : M=128, N=256, K=1024, f32 out
  k_gemm_bf16<<<dim3(1, Ssz / 128, Bsz), 256, 0, stream>>>(
      q_bf, ctx_bf, nullptr, scores, nullptr, Hsz, Hsz, Hsz, Ssz,
      (long)Tsz * Hsz, (long)Ssz * Hsz, (long)Tsz * Ssz);

  k_softmax<<<dim3(Bsz * Tsz), 256, 0, stream>>>(scores, slen, attn_bf);

  // context = attn @ ctx per batch : M=128, N=1024, K=256, bf16 out (B-op = ctxT)
  k_gemm_bf16<<<dim3(1, Hsz / 128, Bsz), 256, 0, stream>>>(
      attn_bf, ctxT_bf, nullptr, nullptr, cvec_bf, Ssz, Ssz, Ssz, Hsz,
      (long)Tsz * Ssz, (long)Hsz * Ssz, (long)Tsz * Hsz);

  // h_tilde = tanh([context, lstm_out] @ W_out^T) -> d_out
  k_htilde<<<dim3(Bsz * Tsz / 128, Hsz / 128), 256, 0, stream>>>(
      cvec_bf, lstm_bf, Wout_bf, out);

  k_finalize<<<dim3(2 * Bsz * Hsz / 256), 256, 0, stream>>>(h_st, c_st, out);
}
